// PyramidResidualMoE_83116207112891
// MI455X (gfx1250) — matrix operand order repaired and checked by reading, not compiled, not run on hardware
//
#include <hip/hip_runtime.h>
#include <stdint.h>

typedef __attribute__((ext_vector_type(16))) __bf16 v16bf;
typedef __attribute__((ext_vector_type(8)))  float  v8f;

#define N_EMBD   768
#define NEXP     8
#define NTOK     16384
#define MT       32          // tokens per block tile
#define XSTRIDE  776         // padded LDS row stride (bf16 elems) for X tile
#define ASTRIDE  72          // padded LDS row stride (bf16 elems) for act tile

// expert hidden sizes (mirrors the logspace pyramid, rounded up to 64)
__device__ __constant__ int c_H[NEXP] = {1536, 1920, 2304, 2816, 3392, 4160, 5056, 6144};
static const int h_H[NEXP]            = {1536, 1920, 2304, 2816, 3392, 4160, 5056, 6144};
#define HSUM 27328

// ---------- helpers ----------
__device__ __forceinline__ unsigned short f2bf(float f) {
    unsigned int u = __float_as_uint(f);
    unsigned int r = u + 0x7FFFu + ((u >> 16) & 1u);   // round-to-nearest-even
    return (unsigned short)(r >> 16);
}

// A fragment (16x32 bf16): lane holds m = lane&15; k = k0..k0+15
// caller passes k0 = kk + ((lane&16)?16:0)
__device__ __forceinline__ v16bf ldfragA(const unsigned short* row, int k0) {
    union { v16bf v; uint4 q[2]; } f;
    f.q[0] = *reinterpret_cast<const uint4*>(row + k0);
    f.q[1] = *reinterpret_cast<const uint4*>(row + k0 + 8);
    return f.v;
}

// B fragment (32x16 bf16): lane holds n = lane&15; 16 consecutive k starting at
// kk + ((lane&16)?16:0). caller passes p = rowbase + that offset.
__device__ __forceinline__ v16bf ldfragB(const unsigned short* p) {
    union { v16bf v; uint4 q[2]; } f;
    f.q[0] = *reinterpret_cast<const uint4*>(p);
    f.q[1] = *reinterpret_cast<const uint4*>(p + 8);
    return f.v;
}

__device__ __forceinline__ v8f wmma_bf16(v16bf a, v16bf b, v8f c) {
    return __builtin_amdgcn_wmma_f32_16x16x32_bf16(false, a, false, b, (short)0, c, false, false);
}

// ---------- kernels ----------
__global__ void __launch_bounds__(256) conv_f32_to_bf16(const float* __restrict__ src,
                                                        unsigned short* __restrict__ dst, int n) {
    for (int i = blockIdx.x * 256 + threadIdx.x; i < n; i += gridDim.x * 256)
        dst[i] = f2bf(src[i]);
}

// one wave per token: router logits -> softmax -> top2 -> scatter into expert lists
__global__ void __launch_bounds__(256) router_kernel(const float* __restrict__ x,
                                                     const float* __restrict__ rW,
                                                     int* __restrict__ counts,
                                                     int* __restrict__ lists,
                                                     float* __restrict__ wlists) {
    const int lane = threadIdx.x & 31;
    const int wv   = threadIdx.x >> 5;
    const int t    = blockIdx.x * 8 + wv;
    if (t >= NTOK) return;
    const float* xr = x + (size_t)t * N_EMBD;
    float acc[NEXP];
#pragma unroll
    for (int e = 0; e < NEXP; ++e) acc[e] = 0.f;
    for (int c = lane; c < N_EMBD; c += 32) {
        float xv = xr[c];
#pragma unroll
        for (int e = 0; e < NEXP; ++e) acc[e] += xv * rW[e * N_EMBD + c];
    }
#pragma unroll
    for (int e = 0; e < NEXP; ++e) {
#pragma unroll
        for (int off = 16; off > 0; off >>= 1)
            acc[e] += __shfl_xor(acc[e], off, 32);
    }
    if (lane == 0) {
        float mx = acc[0];
#pragma unroll
        for (int e = 1; e < NEXP; ++e) mx = fmaxf(mx, acc[e]);
        float p[NEXP]; float s = 0.f;
#pragma unroll
        for (int e = 0; e < NEXP; ++e) { p[e] = __expf(acc[e] - mx); s += p[e]; }
        float inv = 1.f / s;
#pragma unroll
        for (int e = 0; e < NEXP; ++e) p[e] *= inv;
        int i1 = 0; float v1 = p[0];
#pragma unroll
        for (int e = 1; e < NEXP; ++e) if (p[e] > v1) { v1 = p[e]; i1 = e; }
        int i2 = (i1 == 0) ? 1 : 0; float v2 = p[i2];
#pragma unroll
        for (int e = 0; e < NEXP; ++e) if (e != i1 && p[e] > v2) { v2 = p[e]; i2 = e; }
        int pos1 = atomicAdd(&counts[i1], 1);
        lists[i1 * NTOK + pos1] = t; wlists[i1 * NTOK + pos1] = v1;
        int pos2 = atomicAdd(&counts[i2], 1);
        lists[i2 * NTOK + pos2] = t; wlists[i2 * NTOK + pos2] = v2;
    }
}

// fused grouped expert FFN: gate/up GEMM + silu*mul + down GEMM, all bf16 WMMA
__global__ void __launch_bounds__(256) moe_kernel(float* __restrict__ out,
                                                  const unsigned short* __restrict__ wsbf,
                                                  const int* __restrict__ counts,
                                                  const int* __restrict__ lists,
                                                  const float* __restrict__ wlists) {
    __shared__ unsigned short XT[MT * XSTRIDE];    // 32 x 768 bf16, padded
    __shared__ unsigned short ACT[MT * ASTRIDE];   // 32 x 64 bf16, padded
    __shared__ int   s_tok[MT];
    __shared__ float s_w[MT];

    const int e  = blockIdx.y;
    const int ne = counts[e];
    const int m0 = blockIdx.x * MT;
    if (m0 >= ne) return;

    const int tid  = threadIdx.x;
    const int lane = tid & 31;
    const int wv   = tid >> 5;
    const int he   = c_H[e];

    size_t woff = (size_t)NTOK * N_EMBD;
    for (int i = 0; i < e; ++i) woff += 3ull * (size_t)c_H[i] * N_EMBD;
    const unsigned short* gateW = wsbf + woff;                       // [he, 768]
    const unsigned short* upW   = gateW + (size_t)he * N_EMBD;       // [he, 768]
    const unsigned short* downW = upW   + (size_t)he * N_EMBD;       // [768, he]

    if (tid < MT) {
        int gi = m0 + tid;
        if (gi < ne) { s_tok[tid] = lists[e * NTOK + gi]; s_w[tid] = wlists[e * NTOK + gi]; }
        else         { s_tok[tid] = 0;                    s_w[tid] = 0.f; }
    }
    __syncthreads();

    // gather X tile (bf16) -> LDS, 16B chunks
    for (int i = tid; i < MT * (N_EMBD / 8); i += 256) {
        int r  = i / (N_EMBD / 8);
        int cq = i % (N_EMBD / 8);
        const uint4* src = reinterpret_cast<const uint4*>(wsbf + (size_t)s_tok[r] * N_EMBD) + cq;
        reinterpret_cast<uint4*>(&XT[r * XSTRIDE])[cq] = *src;
    }
    __syncthreads();

    const int khA = (lane & 16) ? 16 : 0;    // A-fragment k offset
    const int khB = (lane & 16) ? 16 : 0;    // B-fragment k offset
    const int nlo = lane & 15;

    v8f zero = {0.f, 0.f, 0.f, 0.f, 0.f, 0.f, 0.f, 0.f};
    v8f dacc[2][6];
#pragma unroll
    for (int a = 0; a < 2; ++a)
#pragma unroll
        for (int b = 0; b < 6; ++b) dacc[a][b] = zero;

    const int mtg = wv >> 2;     // 0..1 : which 16-token half (gate/up phase)
    const int ntg = wv & 3;      // 0..3 : which 16 of 64 h-cols (gate/up phase)
    const unsigned short* arowX = &XT[(mtg * 16 + nlo) * XSTRIDE];

    for (int hc = 0; hc < he; hc += 64) {
        // ---- gate & up: this wave's 16x16 tile of the [32 x 64] chunk ----
        const int hrow = hc + ntg * 16 + nlo;
        const unsigned short* grow = gateW + (size_t)hrow * N_EMBD;
        const unsigned short* urow = upW   + (size_t)hrow * N_EMBD;
        v8f gacc = zero, uacc = zero;
        for (int kk = 0; kk < N_EMBD; kk += 32) {
            v16bf a  = ldfragA(arowX, kk + khA);
            v16bf bg = ldfragB(grow + kk + khB);
            v16bf bu = ldfragB(urow + kk + khB);
            gacc = wmma_bf16(a, bg, gacc);
            uacc = wmma_bf16(a, bu, uacc);
        }
        // act = silu(g) * u  -> LDS (bf16)
        const int mbase = mtg * 16 + ((lane & 16) ? 8 : 0);
        const int nb    = ntg * 16 + nlo;
#pragma unroll
        for (int r = 0; r < 8; ++r) {
            float g   = gacc[r];
            float act = (g / (1.f + __expf(-g))) * uacc[r];
            ACT[(mbase + r) * ASTRIDE + nb] = f2bf(act);
        }
        __syncthreads();
        // ---- down-projection accumulate: wave owns 6 of 48 output N-tiles ----
#pragma unroll
        for (int mtd = 0; mtd < 2; ++mtd) {
            const unsigned short* arowA = &ACT[(mtd * 16 + nlo) * ASTRIDE];
            v16bf a0 = ldfragA(arowA, 0 + khA);
            v16bf a1 = ldfragA(arowA, 32 + khA);
#pragma unroll
            for (int j = 0; j < 6; ++j) {
                const int cg = (wv * 6 + j) * 16 + nlo;            // out channel
                const unsigned short* drow = downW + (size_t)cg * he + hc;
                v16bf b0 = ldfragB(drow + khB);
                v16bf b1 = ldfragB(drow + 32 + khB);
                dacc[mtd][j] = wmma_bf16(a0, b0, dacc[mtd][j]);
                dacc[mtd][j] = wmma_bf16(a1, b1, dacc[mtd][j]);
            }
        }
        __syncthreads();
    }

    // writeback: out[t, c] += w_t * y  (token belongs to exactly 2 experts)
#pragma unroll
    for (int mtd = 0; mtd < 2; ++mtd) {
        const int mb = mtd * 16 + ((lane & 16) ? 8 : 0);
#pragma unroll
        for (int j = 0; j < 6; ++j) {
            const int cg = (wv * 6 + j) * 16 + nlo;
#pragma unroll
            for (int r = 0; r < 8; ++r) {
                const int m = mb + r;
                const float wgt = s_w[m];
                if (wgt != 0.f)
                    atomicAdd(&out[(size_t)s_tok[m] * N_EMBD + cg], wgt * dacc[mtd][j][r]);
            }
        }
    }
}

// ---------- host ----------
extern "C" void kernel_launch(void* const* d_in, const int* in_sizes, int n_in,
                              void* d_out, int out_size, void* d_ws, size_t ws_size,
                              hipStream_t stream) {
    (void)in_sizes; (void)n_in; (void)out_size; (void)ws_size;
    const float* x  = (const float*)d_in[0];
    const float* rW = (const float*)d_in[1];
    float* out = (float*)d_out;

    // workspace layout: [bf16 x | bf16 weights (gate,up,down per expert)] [counts][lists][wlists]
    unsigned short* wsbf = (unsigned short*)d_ws;
    size_t bfElems = (size_t)NTOK * N_EMBD + 3ull * N_EMBD * HSUM;
    char* p = (char*)d_ws + bfElems * 2;
    int*   counts = (int*)p;   p += 256;
    int*   lists  = (int*)p;   p += sizeof(int) * (size_t)NEXP * NTOK;
    float* wlists = (float*)p;

    // residual: out = x
    hipMemcpyAsync(out, x, (size_t)NTOK * N_EMBD * sizeof(float),
                   hipMemcpyDeviceToDevice, stream);

    // convert x to bf16
    conv_f32_to_bf16<<<4096, 256, 0, stream>>>(x, wsbf, NTOK * N_EMBD);

    // convert weights to bf16 (dict order: gate_e, up_e, down_e interleaved)
    size_t off = (size_t)NTOK * N_EMBD;
    for (int e = 0; e < NEXP; ++e) {
        int nw = h_H[e] * N_EMBD;
        int blocks = (nw + 255) / 256; if (blocks > 2048) blocks = 2048;
        conv_f32_to_bf16<<<blocks, 256, 0, stream>>>((const float*)d_in[2 + 3 * e], wsbf + off, nw); off += nw;
        conv_f32_to_bf16<<<blocks, 256, 0, stream>>>((const float*)d_in[3 + 3 * e], wsbf + off, nw); off += nw;
        conv_f32_to_bf16<<<blocks, 256, 0, stream>>>((const float*)d_in[4 + 3 * e], wsbf + off, nw); off += nw;
    }

    hipMemsetAsync(counts, 0, NEXP * sizeof(int), stream);
    router_kernel<<<NTOK / 8, 256, 0, stream>>>(x, rW, counts, lists, wlists);

    dim3 grid(NTOK / MT, NEXP);   // 512 token-tiles x 8 experts, early-exit on count
    moe_kernel<<<grid, 256, 0, stream>>>(out, wsbf, counts, lists, wlists);
}
